// GPT2Model_31679678775662
// MI455X (gfx1250) — matrix operand order repaired and checked by reading, not compiled, not run on hardware
//
#include <hip/hip_runtime.h>
#include <hip/hip_bf16.h>

#define DEV_INLINE __device__ __forceinline__

typedef __attribute__((ext_vector_type(16))) __bf16 v16bf;
typedef __attribute__((ext_vector_type(8)))  float  v8f;
typedef __attribute__((ext_vector_type(4)))  int    v4i;

// Probe CDNA5 async global->LDS path (ASYNCcnt); fall back to uint4 copies.
#if defined(__has_builtin)
#  if __has_builtin(__builtin_amdgcn_global_load_async_to_lds_b128)
#    define HAVE_ASYNC_LDS 1
#  endif
#  if __has_builtin(__builtin_amdgcn_s_wait_asynccnt)
#    define HAVE_WAIT_ASYNC 1
#  endif
#endif
#ifndef HAVE_ASYNC_LDS
#  define HAVE_ASYNC_LDS 0
#endif
#ifndef HAVE_WAIT_ASYNC
#  define HAVE_WAIT_ASYNC 0
#endif

// Model dims (fixed by the reference)
constexpr int Dm = 1024;    // n_embd
constexpr int Sm = 512;     // seq
constexpr int Bm = 4;       // batch
constexpr int Hm = 16;      // heads
constexpr int Lm = 12;      // layers
constexpr int Vm = 16386;   // vocab
constexpr int Mm = Bm * Sm; // 2048 rows

// ---------------------------------------------------------------------------
// Async copy helpers
// ---------------------------------------------------------------------------
DEV_INLINE void copy_b128(const void* g, void* l) {
#if HAVE_ASYNC_LDS
  __builtin_amdgcn_global_load_async_to_lds_b128(
      (__attribute__((address_space(1))) v4i*)(g),
      (__attribute__((address_space(3))) v4i*)(l), 0, 0);
#else
  *(uint4*)l = *(const uint4*)g;
#endif
}

DEV_INLINE void copy_fence() {
#if HAVE_ASYNC_LDS
#  if HAVE_WAIT_ASYNC
  __builtin_amdgcn_s_wait_asynccnt(0);
#  else
  asm volatile("s_wait_asynccnt 0x0" ::: "memory");
#  endif
#endif
}

// ---------------------------------------------------------------------------
// WMMA helpers (CDNA5 v_wmma_f32_16x16x32_bf16, contiguous K per lane half for A and B)
// ---------------------------------------------------------------------------
DEV_INLINE v8f wmma_bf16(v16bf a, v16bf b, v8f c) {
  return __builtin_amdgcn_wmma_f32_16x16x32_bf16(
      /*neg_a=*/false, a, /*neg_b=*/false, b,
      /*c_mod=*/(short)0, c, /*reuse_a=*/false, /*reuse_b=*/false);
}

// A fragment: 16x32 bf16 tile, row-major in LDS, leading dim ld.
// lane<16: row=lane,    K = 0..15
// lane>=16: row=lane-16, K = 16..31
DEV_INLINE v16bf frag_a(const __bf16* tile, int ld, int lane) {
  const __bf16* r = tile + (lane & 15) * ld + ((lane >> 4) & 1) * 16;
  v16bf f;
#pragma unroll
  for (int i = 0; i < 8; ++i) f[i] = r[i];
#pragma unroll
  for (int i = 0; i < 8; ++i) f[8 + i] = r[8 + i];
  return f;
}

// B fragment from K-major tile (tile[k*ld + n]), 32x16.
// lane<16: col=lane, K=0..15 ; lane>=16: col=lane-16, K=16..31
DEV_INLINE v16bf frag_b_kmajor(const __bf16* tile, int ld, int lane) {
  const __bf16* c = tile + (lane & 15) + ((lane >> 4) & 1) * 16 * ld;
  v16bf f;
#pragma unroll
  for (int i = 0; i < 16; ++i) f[i] = c[i * ld];
  return f;
}

// B fragment from N-major tile (tile[n*ld + k]) — used for Q·K^T.
DEV_INLINE v16bf frag_b_nmajor(const __bf16* tile, int ld, int lane) {
  const __bf16* r = tile + (lane & 15) * ld + ((lane >> 4) & 1) * 16;
  v16bf f;
#pragma unroll
  for (int i = 0; i < 16; ++i) f[i] = r[i];
  return f;
}

DEV_INLINE float gelu_new(float x) {
  return 0.5f * x * (1.0f + tanhf(0.7978845608028654f * (x + 0.044715f * x * x * x)));
}

// ---------------------------------------------------------------------------
// Embedding: h[b,s,:] = wte[id] + wpe[s] + wtte[tt]
// ---------------------------------------------------------------------------
__global__ __launch_bounds__(256) void embed_kernel(
    const int* __restrict__ ids, const int* __restrict__ tts,
    const float* __restrict__ wte, const float* __restrict__ wtte,
    const float* __restrict__ wpe, float* __restrict__ h) {
  int tok = blockIdx.x;
  int s = tok & (Sm - 1);
  int id = ids[tok];
  int tt = tts[tok];
  int t = threadIdx.x;
#pragma unroll
  for (int i = 0; i < 4; ++i) {
    int d = t + i * 256;
    h[(size_t)tok * Dm + d] = wte[(size_t)id * Dm + d] +
                              wpe[(size_t)s * Dm + d] +
                              wtte[(size_t)tt * Dm + d];
  }
}

// ---------------------------------------------------------------------------
// LayerNorm over D=1024, writes bf16 (GEMM A operand)
// ---------------------------------------------------------------------------
__global__ __launch_bounds__(256) void ln_kernel(
    const float* __restrict__ X, const float* __restrict__ g,
    const float* __restrict__ b, __bf16* __restrict__ Y) {
  int row = blockIdx.x;
  int t = threadIdx.x;
  const float* x = X + (size_t)row * Dm;
  float v[4];
  float s = 0.f;
#pragma unroll
  for (int i = 0; i < 4; ++i) { v[i] = x[t + i * 256]; s += v[i]; }
  __shared__ float red[256];
  red[t] = s;
  __syncthreads();
  for (int off = 128; off > 0; off >>= 1) {
    if (t < off) red[t] += red[t + off];
    __syncthreads();
  }
  float mu = red[0] * (1.0f / Dm);
  __syncthreads();
  float s2 = 0.f;
#pragma unroll
  for (int i = 0; i < 4; ++i) { float d = v[i] - mu; s2 += d * d; }
  red[t] = s2;
  __syncthreads();
  for (int off = 128; off > 0; off >>= 1) {
    if (t < off) red[t] += red[t + off];
    __syncthreads();
  }
  float rstd = rsqrtf(red[0] * (1.0f / Dm) + 1e-5f);
#pragma unroll
  for (int i = 0; i < 4; ++i) {
    int d = t + i * 256;
    Y[(size_t)row * Dm + d] = (__bf16)((v[i] - mu) * rstd * g[d] + b[d]);
  }
}

// ---------------------------------------------------------------------------
// Tiled WMMA GEMM: C[M,N] = act(A_bf16[M,K] @ W_f32->bf16[K,N] + bias) (+Res)
// Block tile 64x128, 8 waves (4 M-waves x 2 N-waves), each wave 16x64 output
// (4 accumulators -> 4 WMMAs per 32-wide K step).
// ---------------------------------------------------------------------------
template <int ACT, bool HAS_RES, bool OUT_F32, bool OUT_BF16>
__global__ __launch_bounds__(256) void gemm_kernel(
    const __bf16* __restrict__ A, const float* __restrict__ W,
    const float* __restrict__ bias, const float* __restrict__ Res,
    float* __restrict__ Cf, __bf16* __restrict__ Cb,
    int Mr, int N, int K) {
  __shared__ __bf16 As[64 * 32];    // 4 KB
  __shared__ __bf16 Ws[32 * 128];   // 8 KB
  const int t = threadIdx.x;
  const int lane = t & 31;
  const int wave = t >> 5;
  const int wm = wave & 3;   // M sub-tile (16 rows)
  const int wn = wave >> 2;  // N sub-tile (64 cols)
  const int m0 = blockIdx.y * 64;
  const int n0 = blockIdx.x * 128;

  v8f acc[4] = {};

  const int arow = t >> 2;        // 0..63
  const int acol = (t & 3) * 8;   // 0,8,16,24 (8 bf16 = 16B per thread)
  const int wrow = t >> 3;        // 0..31
  const int wcol = (t & 7) * 16;  // 16 floats per thread

  for (int k0 = 0; k0 < K; k0 += 32) {
    // Stage A tile 64x32 (bf16 raw copy: async-to-LDS when available)
    copy_b128(A + (size_t)(m0 + arow) * K + k0 + acol, As + arow * 32 + acol);

    // Stage W tile 32x128 (f32 global -> bf16 LDS), zero-pad past N
    {
      const float* src = W + (size_t)(k0 + wrow) * N + n0 + wcol;
      __bf16* dst = Ws + wrow * 128 + wcol;
      if (n0 + wcol + 15 < N) {
        const float2* s2 = (const float2*)src;  // rows are 8B-aligned even for N=16386
#pragma unroll
        for (int i = 0; i < 8; ++i) {
          float2 v = s2[i];
          dst[2 * i + 0] = (__bf16)v.x;
          dst[2 * i + 1] = (__bf16)v.y;
        }
      } else {
#pragma unroll
        for (int i = 0; i < 16; ++i) {
          float v = (n0 + wcol + i < N) ? src[i] : 0.0f;
          dst[i] = (__bf16)v;
        }
      }
      if (k0 + 32 < K)
        __builtin_prefetch(W + (size_t)(k0 + 32 + wrow) * N + n0 + wcol, 0, 0);
    }
    copy_fence();
    __syncthreads();

    v16bf af = frag_a(As + wm * 16 * 32, 32, lane);
#pragma unroll
    for (int j = 0; j < 4; ++j) {
      v16bf bf = frag_b_kmajor(Ws + wn * 64 + j * 16, 128, lane);
      acc[j] = wmma_bf16(af, bf, acc[j]);
    }
    __syncthreads();
  }

  // Epilogue: C layout — lane<16: M=e, N=lane ; lane>=16: M=e+8, N=lane-16
  const int half = (lane >> 4) & 1;
  const int nl = lane & 15;
#pragma unroll
  for (int e = 0; e < 8; ++e) {
    int m = m0 + wm * 16 + e + 8 * half;
#pragma unroll
    for (int j = 0; j < 4; ++j) {
      int n = n0 + wn * 64 + j * 16 + nl;
      if (n < N) {
        float v = acc[j][e] + bias[n];
        if (ACT == 1) v = gelu_new(v);
        if (HAS_RES) v += Res[(size_t)m * N + n];
        if (OUT_F32) Cf[(size_t)m * N + n] = v;
        if (OUT_BF16) Cb[(size_t)m * N + n] = (__bf16)v;
      }
    }
  }
  (void)Mr;
}

// ---------------------------------------------------------------------------
// Flash attention: one wave per (b, h, 16-row q tile). Online softmax.
// qkv: [M, 3D] bf16 (q | k | v per row).  out: [M, D] bf16 (head-merged).
// ---------------------------------------------------------------------------
__global__ __launch_bounds__(32) void attn_kernel(
    const __bf16* __restrict__ qkv, __bf16* __restrict__ out) {
  const int qt = blockIdx.x;
  const int h = blockIdx.y;
  const int bb = blockIdx.z;
  const int lane = threadIdx.x;
  const int qbase = qt * 16;
  const size_t rowbase = (size_t)bb * Sm;

  __shared__ __bf16 Qs[16 * 64];
  __shared__ __bf16 Ks[32 * 64];
  __shared__ __bf16 Vs[32 * 64];
  __shared__ __bf16 Ps[16 * 32];

  // Load Q tile (16 x 64): 128 chunks of 8 bf16 (16B)
#pragma unroll
  for (int i = lane; i < 128; i += 32) {
    int r = i >> 3, c = (i & 7) * 8;
    copy_b128(qkv + (rowbase + qbase + r) * (3 * Dm) + h * 64 + c,
              Qs + r * 64 + c);
  }

  v8f o0 = {}, o1 = {}, o2 = {}, o3 = {};
  float rmax[8], rsum[8];
#pragma unroll
  for (int e = 0; e < 8; ++e) { rmax[e] = -1e30f; rsum[e] = 0.f; }

  const int half = lane >> 4;
  const int nl = lane & 15;
  const float scale = 0.125f;  // 1/sqrt(64)

  for (int kb = 0; kb <= qbase + 15; kb += 32) {
    // Stage K,V chunks (32 keys x 64): 256 chunks of 16B each
#pragma unroll
    for (int i = lane; i < 256; i += 32) {
      int r = i >> 3, c = (i & 7) * 8;
      size_t g = (rowbase + kb + r) * (3 * Dm) + h * 64 + c;
      copy_b128(qkv + g + Dm, Ks + r * 64 + c);
      copy_b128(qkv + g + 2 * Dm, Vs + r * 64 + c);
    }
    copy_fence();
    __syncthreads();

    // Scores: S = Q(16x64) @ K^T(64x32) -> two 16x16 tiles
    v8f s0 = {}, s1 = {};
#pragma unroll
    for (int ks = 0; ks < 64; ks += 32) {
      v16bf qa = frag_a(Qs + ks, 64, lane);
      v16bf kf0 = frag_b_nmajor(Ks + ks, 64, lane);
      v16bf kf1 = frag_b_nmajor(Ks + 16 * 64 + ks, 64, lane);
      s0 = wmma_bf16(qa, kf0, s0);
      s1 = wmma_bf16(qa, kf1, s1);
    }

    // Mask + online softmax update
#pragma unroll
    for (int e = 0; e < 8; ++e) {
      int qi = qbase + e + 8 * half;
      float v0 = s0[e] * scale;
      if (kb + nl > qi) v0 = -10000.0f;
      float v1 = s1[e] * scale;
      if (kb + 16 + nl > qi) v1 = -10000.0f;
      float r = fmaxf(v0, v1);
#pragma unroll
      for (int mk = 1; mk < 16; mk <<= 1) r = fmaxf(r, __shfl_xor(r, mk, 32));
      float nm = fmaxf(rmax[e], r);
      float p0 = __expf(v0 - nm);
      float p1 = __expf(v1 - nm);
      float ps = p0 + p1;
#pragma unroll
      for (int mk = 1; mk < 16; mk <<= 1) ps += __shfl_xor(ps, mk, 32);
      float corr = __expf(rmax[e] - nm);
      rmax[e] = nm;
      rsum[e] = rsum[e] * corr + ps;
      o0[e] *= corr; o1[e] *= corr; o2[e] *= corr; o3[e] *= corr;
      Ps[(e + 8 * half) * 32 + nl] = (__bf16)p0;
      Ps[(e + 8 * half) * 32 + 16 + nl] = (__bf16)p1;
    }
    __syncthreads();

    // O += P(16x32) @ V(32x64)
    v16bf pa = frag_a(Ps, 32, lane);
    o0 = wmma_bf16(pa, frag_b_kmajor(Vs + 0, 64, lane), o0);
    o1 = wmma_bf16(pa, frag_b_kmajor(Vs + 16, 64, lane), o1);
    o2 = wmma_bf16(pa, frag_b_kmajor(Vs + 32, 64, lane), o2);
    o3 = wmma_bf16(pa, frag_b_kmajor(Vs + 48, 64, lane), o3);
    __syncthreads();
  }

  // Normalize and write (merge heads back to [M, D])
#pragma unroll
  for (int e = 0; e < 8; ++e) {
    float inv = 1.0f / rsum[e];
    int m = e + 8 * half;
    size_t rowp = (rowbase + qbase + m) * Dm + h * 64;
    out[rowp + 0 + nl]  = (__bf16)(o0[e] * inv);
    out[rowp + 16 + nl] = (__bf16)(o1[e] * inv);
    out[rowp + 32 + nl] = (__bf16)(o2[e] * inv);
    out[rowp + 48 + nl] = (__bf16)(o3[e] * inv);
  }
}

// ---------------------------------------------------------------------------
// Host orchestration
// ---------------------------------------------------------------------------
extern "C" void kernel_launch(void* const* d_in, const int* in_sizes, int n_in,
                              void* d_out, int out_size, void* d_ws, size_t ws_size,
                              hipStream_t stream) {
  (void)in_sizes; (void)n_in; (void)out_size;
  const int*   input_ids = (const int*)d_in[0];
  const int*   ttids     = (const int*)d_in[1];
  const float* wte   = (const float*)d_in[2];
  const float* wtte  = (const float*)d_in[3];
  const float* wpe   = (const float*)d_in[4];
  const float* ln1w  = (const float*)d_in[5];
  const float* ln1b  = (const float*)d_in[6];
  const float* attnw = (const float*)d_in[7];
  const float* attnb = (const float*)d_in[8];
  const float* atpw  = (const float*)d_in[9];
  const float* atpb  = (const float*)d_in[10];
  const float* ln2w  = (const float*)d_in[11];
  const float* ln2b  = (const float*)d_in[12];
  const float* fcw   = (const float*)d_in[13];
  const float* fcb   = (const float*)d_in[14];
  const float* mlpw  = (const float*)d_in[15];
  const float* mlpb  = (const float*)d_in[16];
  const float* lnfw  = (const float*)d_in[17];
  const float* lnfb  = (const float*)d_in[18];
  const float* headw = (const float*)d_in[19];
  const float* headb = (const float*)d_in[20];
  float* logits = (float*)d_out;

  // Persistent small scratch in d_ws: h (fp32) + LN output (bf16)
  const size_t h_bytes  = (size_t)Mm * Dm * sizeof(float);   // 8 MB
  const size_t xb_bytes = (size_t)Mm * Dm * sizeof(__bf16);  // 4 MB
  if (ws_size < h_bytes + xb_bytes) return;
  char* wsp = (char*)d_ws;
  float*  h  = (float*)wsp;              wsp += h_bytes;
  __bf16* xb = (__bf16*)wsp;             wsp += xb_bytes;

  // Large transients live inside d_out (134 MB; dead until final head GEMM)
  char* op = (char*)d_out;
  __bf16* qkvb  = (__bf16*)op;                                   // 12.6 MB
  __bf16* attb  = (__bf16*)(op + (size_t)Mm * 3 * Dm * 2);       //  4.2 MB
  __bf16* fcbuf = (__bf16*)(op + (size_t)Mm * 4 * Dm * 2);       // 16.8 MB

  embed_kernel<<<Mm, 256, 0, stream>>>(input_ids, ttids, wte, wtte, wpe, h);

  for (int l = 0; l < Lm; ++l) {
    ln_kernel<<<Mm, 256, 0, stream>>>(h, ln1w + l * Dm, ln1b + l * Dm, xb);
    gemm_kernel<0, false, false, true><<<dim3(3 * Dm / 128, Mm / 64), 256, 0, stream>>>(
        xb, attnw + (size_t)l * Dm * 3 * Dm, attnb + (size_t)l * 3 * Dm,
        nullptr, nullptr, qkvb, Mm, 3 * Dm, Dm);
    attn_kernel<<<dim3(Sm / 16, Hm, Bm), 32, 0, stream>>>(qkvb, attb);
    gemm_kernel<0, true, true, false><<<dim3(Dm / 128, Mm / 64), 256, 0, stream>>>(
        attb, atpw + (size_t)l * Dm * Dm, atpb + (size_t)l * Dm,
        h, h, nullptr, Mm, Dm, Dm);
    ln_kernel<<<Mm, 256, 0, stream>>>(h, ln2w + l * Dm, ln2b + l * Dm, xb);
    gemm_kernel<1, false, false, true><<<dim3(4 * Dm / 128, Mm / 64), 256, 0, stream>>>(
        xb, fcw + (size_t)l * Dm * 4 * Dm, fcb + (size_t)l * 4 * Dm,
        nullptr, nullptr, fcbuf, Mm, 4 * Dm, Dm);
    gemm_kernel<0, true, true, false><<<dim3(Dm / 128, Mm / 64), 256, 0, stream>>>(
        fcbuf, mlpw + (size_t)l * 4 * Dm * Dm, mlpb + (size_t)l * Dm,
        h, h, nullptr, Mm, Dm, 4 * Dm);
  }

  ln_kernel<<<Mm, 256, 0, stream>>>(h, lnfw, lnfb, xb);
  gemm_kernel<0, false, true, false><<<dim3((Vm + 127) / 128, Mm / 64), 256, 0, stream>>>(
      xb, headw, headb, nullptr, logits, nullptr, Mm, Vm, Dm);
}
